// CAN_Layer_74775380623980
// MI455X (gfx1250) — matrix operand order repaired and checked by reading, not compiled, not run on hardware
//
#include <hip/hip_runtime.h>
#include <hip/hip_bf16.h>

// Reference collapses: seq len == 1 -> softmax over one key -> alpha == 1.
// out = [E | E] with E = 0.5*(protein @ Wv_p + drug @ Wv_d)  (16384 x 4096 f32)
// One fused GEMM: X=[protein|drug] (16384x4096) @ W=[Wv_p;Wv_d] (4096x2048), K=4096.
// Prepass converts X and W to bf16 (W pre-tiled to LDS panel order) so the
// GEMM hot loop is pure async DMA + WMMA with no conversion VALU.
// Wave tile 32x64 (64-VGPR acc) + batch fragment loads: ~140 live VGPRs so the
// allocator keeps accumulators pinned (no v_mov_b64 rotation, no spills).

typedef __bf16 bf16;
typedef bf16  v16bf __attribute__((ext_vector_type(16)));
typedef float v8f   __attribute__((ext_vector_type(8)));
typedef int   v4i   __attribute__((ext_vector_type(4)));
typedef unsigned short u16;
typedef unsigned int   u32;

#define NROWS 16384
#define DIN   2048
#define HID   2048
#define KTOT  4096
#define BM    128
#define BN    128
#define BK    32
#define LDSS  40               // padded LDS row stride (bf16); 80 B rows, 16B aligned
#define KT    (KTOT / BK)      // 128
#define PANEL (BN * BK)        // 4096 bf16 = 8 KB per (nTile, kt) weight panel

#if __has_builtin(__builtin_amdgcn_global_load_async_to_lds_b128) && \
    __has_builtin(__builtin_amdgcn_s_wait_asynccnt)
#define HAVE_ASYNC 1
typedef __attribute__((address_space(1))) v4i* gv4i_p;   // global src, b128-typed
typedef __attribute__((address_space(3))) v4i* lv4i_p;   // LDS dst, b128-typed
#else
#define HAVE_ASYNC 0
#endif

__device__ __forceinline__ u32 pack2(float lo, float hi) {
  union { float f; u32 u; } a, b; a.f = lo; b.f = hi;
  u32 ra = (a.u + 0x7FFFu + ((a.u >> 16) & 1u)) >> 16;   // RNE
  u32 rb = (b.u + 0x7FFFu + ((b.u >> 16) & 1u)) >> 16;
  return ra | (rb << 16);
}

// ---------------- prepass: X = [protein | drug] -> bf16 row-major ----------
__global__ __launch_bounds__(256)
void convert_x(const float* __restrict__ prot, const float* __restrict__ drug,
               u16* __restrict__ xb) {
  const int idx = blockIdx.x * 256 + threadIdx.x;        // one thread per 8 elems
  const int row = idx >> 9;                              // KTOT/8 = 512
  const int col = (idx & 511) << 3;
  const float* src = (col < DIN) ? (prot + (size_t)row * DIN + col)
                                 : (drug + (size_t)row * DIN + (col - DIN));
  float4 f0 = ((const float4*)src)[0];
  float4 f1 = ((const float4*)src)[1];
  uint4 q;
  q.x = pack2(f0.x, f0.y); q.y = pack2(f0.z, f0.w);
  q.z = pack2(f1.x, f1.y); q.w = pack2(f1.z, f1.w);
  *(uint4*)(xb + (size_t)row * KTOT + col) = q;
}

// ---- prepass: W = [Wv_p ; Wv_d] -> bf16, pre-tiled to LDS panel order -----
// layout: [nTile=16][kt=128][n'=128][k'=32]; 256 threads = 2 kt panels / block
__global__ __launch_bounds__(256)
void convert_w(const float* __restrict__ wvp, const float* __restrict__ wvd,
               u16* __restrict__ wt) {
  const int nTile = blockIdx.x;                          // 0..15
  const int kt    = blockIdx.y * 2 + (threadIdx.x >> 7); // 0..127
  const int np    = threadIdx.x & 127;                   // n' in panel
  const int n     = nTile * BN + np;
  const int k0    = kt * BK;
  const float* w = (k0 < DIN) ? (wvp + (size_t)k0 * HID + n)
                              : (wvd + (size_t)(k0 - DIN) * HID + n);
  u32 q[16];
  #pragma unroll
  for (int i = 0; i < 16; ++i)                           // coalesced along n
    q[i] = pack2(w[(size_t)(2 * i) * HID], w[(size_t)(2 * i + 1) * HID]);
  u16* dst = wt + ((size_t)(nTile * KT + kt) * BN + np) * BK;
  #pragma unroll
  for (int i = 0; i < 4; ++i)
    ((uint4*)dst)[i] = ((uint4*)q)[i];                   // 64 B contiguous/thread
}

// ---------------- main GEMM: bf16 in, f32 out, async DMA staging -----------
__global__ __launch_bounds__(256)
void gemm_bf16(const u16* __restrict__ xb, const u16* __restrict__ wt,
               float* __restrict__ out) {
  __shared__ __align__(16) u16 sA[2][BM * LDSS];
  __shared__ __align__(16) u16 sB[2][BN * LDSS];

  const int tid   = threadIdx.x;
  const int lane  = tid & 31;
  const int wave  = tid >> 5;       // 8 waves: 4(M) x 2(N), each 32x64
  const int waveM = wave >> 1;      // 0..3
  const int waveN = wave & 1;       // 0..1
  const int bm    = blockIdx.y * BM;
  const int nTile = blockIdx.x;
  const int bn    = nTile * BN;

  // staging: both A and B tiles are 128 rows x 32 bf16; thread -> (row=tid>>1, half=tid&1)
  const u16* gA = xb + (size_t)(bm + (tid >> 1)) * KTOT + (tid & 1) * 16;          // +32/kt
  const u16* gB = wt + (size_t)nTile * KT * PANEL
                     + (size_t)(tid >> 1) * BK + (tid & 1) * 16;                   // +PANEL/kt
  const int lOff = (tid >> 1) * LDSS + (tid & 1) * 16;   // same for sA and sB

#if HAVE_ASYNC
  auto issue = [&](int kt, int buf) {
    gv4i_p ga = (gv4i_p)(void*)(gA + (size_t)kt * BK);
    lv4i_p la = (lv4i_p)(void*)&sA[buf][lOff];
    __builtin_amdgcn_global_load_async_to_lds_b128(ga, la, 0, 0);
    __builtin_amdgcn_global_load_async_to_lds_b128(ga, la, 16, 0);
    gv4i_p gb = (gv4i_p)(void*)(gB + (size_t)kt * PANEL);
    lv4i_p lb = (lv4i_p)(void*)&sB[buf][lOff];
    __builtin_amdgcn_global_load_async_to_lds_b128(gb, lb, 0, 0);
    __builtin_amdgcn_global_load_async_to_lds_b128(gb, lb, 16, 0);
  };
#else
  uint4 ra[2], rbv[2];
  auto loadT = [&](int kt) {
    const uint4* pa = (const uint4*)(gA + (size_t)kt * BK);
    ra[0] = pa[0]; ra[1] = pa[1];
    const uint4* pb = (const uint4*)(gB + (size_t)kt * PANEL);
    rbv[0] = pb[0]; rbv[1] = pb[1];
  };
  auto storeT = [&](int buf) {
    uint4* pa = (uint4*)&sA[buf][lOff];
    pa[0] = ra[0]; pa[1] = ra[1];
    uint4* pb = (uint4*)&sB[buf][lOff];
    pb[0] = rbv[0]; pb[1] = rbv[1];
  };
#endif

  // per-lane fragment offsets (ISA 16-bit A / B VGPR layouts)
  const int lLo = lane & 15;
  const int lHi = lane >> 4;
  int offA[2], offB[4];
  #pragma unroll
  for (int mi = 0; mi < 2; ++mi)
    offA[mi] = (waveM * 32 + mi * 16 + lLo) * LDSS + lHi * 8;
  #pragma unroll
  for (int ni = 0; ni < 4; ++ni)
    offB[ni] = (waveN * 64 + ni * 16 + lLo) * LDSS + lHi * 8;

  v8f acc[2][4];
  #pragma unroll
  for (int mi = 0; mi < 2; ++mi)
    #pragma unroll
    for (int ni = 0; ni < 4; ++ni)
      acc[mi][ni] = (v8f)0.0f;

  // Batch-load all 6 fragments (48 VGPRs), then 8 back-to-back WMMAs.
  auto compute = [&](int buf) {
    const u16* A = sA[buf];
    const u16* B = sB[buf];
    v16bf af[2], bfr[4];
    #pragma unroll
    for (int mi = 0; mi < 2; ++mi) {
      uint4* p = (uint4*)&af[mi];
      p[0] = *(const uint4*)&A[offA[mi]];          // K = lHi*8 .. +7
      p[1] = *(const uint4*)&A[offA[mi] + 16];     // K = lHi*8+16 .. +23
    }
    #pragma unroll
    for (int ni = 0; ni < 4; ++ni) {
      uint4* p = (uint4*)&bfr[ni];
      p[0] = *(const uint4*)&B[offB[ni]];          // K = lHi*8 .. +7
      p[1] = *(const uint4*)&B[offB[ni] + 16];     // K = lHi*8+16 .. +23
    }
    #pragma unroll
    for (int mi = 0; mi < 2; ++mi)
      #pragma unroll
      for (int ni = 0; ni < 4; ++ni)
        acc[mi][ni] = __builtin_amdgcn_wmma_f32_16x16x32_bf16(
            false, af[mi], false, bfr[ni], (short)0, acc[mi][ni], false, false);
  };

#if HAVE_ASYNC
  issue(0, 0);
  for (int kt = 0; kt < KT; ++kt) {
    const int cur = kt & 1;
    __builtin_amdgcn_s_wait_asynccnt(0);           // own async DMAs landed in LDS
    __syncthreads();                               // all waves' tiles visible; buf^1 free
    if (kt + 1 < KT) issue(kt + 1, cur ^ 1);       // DMA next tile under the WMMAs
    compute(cur);
  }
#else
  loadT(0);
  storeT(0);
  __syncthreads();
  for (int kt = 0; kt < KT; ++kt) {
    const int cur = kt & 1;
    if (kt + 1 < KT) loadT(kt + 1);
    compute(cur);
    if (kt + 1 < KT) {
      __syncthreads();
      storeT(cur ^ 1);
      __syncthreads();
    }
  }
#endif

  // epilogue: C layout VGPR r: row = r + 8*(lane>>4), col = lane&15
  #pragma unroll
  for (int mi = 0; mi < 2; ++mi) {
    const int row0 = bm + waveM * 32 + mi * 16 + lHi * 8;
    #pragma unroll
    for (int ni = 0; ni < 4; ++ni) {
      const int col = bn + waveN * 64 + ni * 16 + lLo;
      float* o = out + (size_t)row0 * (2 * HID) + col;
      #pragma unroll
      for (int r = 0; r < 8; ++r) {
        const float v = acc[mi][ni][r] * 0.5f;
        __builtin_nontemporal_store(v, o + (size_t)r * (2 * HID));
        __builtin_nontemporal_store(v, o + (size_t)r * (2 * HID) + HID);
      }
    }
  }
}

// ------------- fallback (ws too small): in-loop conversion GEMM ------------
__global__ __launch_bounds__(256)
void can_fused_gemm(const float* __restrict__ prot, const float* __restrict__ drug,
                    const float* __restrict__ wvp, const float* __restrict__ wvd,
                    float* __restrict__ out) {
  __shared__ __align__(16) u16 sA[2][BM * LDSS];
  __shared__ __align__(16) u16 sB[2][BN * LDSS];
  const int tid = threadIdx.x, lane = tid & 31, wave = tid >> 5;
  const int waveM = wave >> 1, waveN = wave & 1;
  const int bm = blockIdx.y * BM, bn = blockIdx.x * BN;
  const int aRow = tid >> 1, aCol = (tid & 1) * 16;
  const float* aP = prot + (size_t)(bm + aRow) * DIN + aCol;
  const float* aD = drug + (size_t)(bm + aRow) * DIN + aCol;
  const int sAoff = aRow * LDSS + aCol;
  const int kb = (tid >> 5) * 4, nb = (tid & 31) * 4;    // 32k x 128n tile
  const float* bP = wvp + (size_t)kb * HID + bn + nb;
  const float* bD = wvd + (size_t)kb * HID + bn + nb;
  const int sBoff = nb * LDSS + kb;
  float4 ra[4], rb[4];
  auto loadTile = [&](int kt) {
    const int k0 = kt * BK;
    const float *pa, *pb;
    if (k0 < DIN) { pa = aP + k0;         pb = bP + (size_t)k0 * HID; }
    else          { pa = aD + (k0 - DIN); pb = bD + (size_t)(k0 - DIN) * HID; }
    #pragma unroll
    for (int i = 0; i < 4; ++i) ra[i] = *(const float4*)(pa + i * 4);
    #pragma unroll
    for (int i = 0; i < 4; ++i) rb[i] = *(const float4*)(pb + (size_t)i * HID);
  };
  auto storeTile = [&](int buf) {
    uint4 q0, q1;
    q0.x = pack2(ra[0].x, ra[0].y); q0.y = pack2(ra[0].z, ra[0].w);
    q0.z = pack2(ra[1].x, ra[1].y); q0.w = pack2(ra[1].z, ra[1].w);
    q1.x = pack2(ra[2].x, ra[2].y); q1.y = pack2(ra[2].z, ra[2].w);
    q1.z = pack2(ra[3].x, ra[3].y); q1.w = pack2(ra[3].z, ra[3].w);
    *(uint4*)&sA[buf][sAoff]     = q0;
    *(uint4*)&sA[buf][sAoff + 8] = q1;
    const float* rbf = (const float*)rb;                 // rbf[i*4+j]: k=kb+i, n=nb+j
    #pragma unroll
    for (int j = 0; j < 4; ++j) {
      uint2 q;
      q.x = pack2(rbf[0 * 4 + j], rbf[1 * 4 + j]);
      q.y = pack2(rbf[2 * 4 + j], rbf[3 * 4 + j]);
      *(uint2*)&sB[buf][sBoff + j * LDSS] = q;
    }
  };
  const int lLo = lane & 15, lHi = lane >> 4;
  int offA[2], offB[4];
  #pragma unroll
  for (int mi = 0; mi < 2; ++mi)
    offA[mi] = (waveM * 32 + mi * 16 + lLo) * LDSS + lHi * 8;
  #pragma unroll
  for (int ni = 0; ni < 4; ++ni)
    offB[ni] = (waveN * 64 + ni * 16 + lLo) * LDSS + lHi * 8;
  v8f acc[2][4];
  #pragma unroll
  for (int mi = 0; mi < 2; ++mi)
    #pragma unroll
    for (int ni = 0; ni < 4; ++ni) acc[mi][ni] = (v8f)0.0f;
  loadTile(0); storeTile(0); __syncthreads();
  for (int kt = 0; kt < KT; ++kt) {
    const int cur = kt & 1;
    if (kt + 1 < KT) loadTile(kt + 1);
    const u16* A = sA[cur]; const u16* B = sB[cur];
    v16bf af[2], bfr[4];
    #pragma unroll
    for (int mi = 0; mi < 2; ++mi) {
      uint4* p = (uint4*)&af[mi];
      p[0] = *(const uint4*)&A[offA[mi]];
      p[1] = *(const uint4*)&A[offA[mi] + 16];
    }
    #pragma unroll
    for (int ni = 0; ni < 4; ++ni) {
      uint4* p = (uint4*)&bfr[ni];
      p[0] = *(const uint4*)&B[offB[ni]];
      p[1] = *(const uint4*)&B[offB[ni] + 16];
    }
    #pragma unroll
    for (int mi = 0; mi < 2; ++mi)
      #pragma unroll
      for (int ni = 0; ni < 4; ++ni)
        acc[mi][ni] = __builtin_amdgcn_wmma_f32_16x16x32_bf16(
            false, af[mi], false, bfr[ni], (short)0, acc[mi][ni], false, false);
    if (kt + 1 < KT) { __syncthreads(); storeTile(cur ^ 1); __syncthreads(); }
  }
  #pragma unroll
  for (int mi = 0; mi < 2; ++mi) {
    const int row0 = bm + waveM * 32 + mi * 16 + lHi * 8;
    #pragma unroll
    for (int ni = 0; ni < 4; ++ni) {
      const int col = bn + waveN * 64 + ni * 16 + lLo;
      float* o = out + (size_t)row0 * (2 * HID) + col;
      #pragma unroll
      for (int r = 0; r < 8; ++r) {
        const float v = acc[mi][ni][r] * 0.5f;
        __builtin_nontemporal_store(v, o + (size_t)r * (2 * HID));
        __builtin_nontemporal_store(v, o + (size_t)r * (2 * HID) + HID);
      }
    }
  }
}

extern "C" void kernel_launch(void* const* d_in, const int* in_sizes, int n_in,
                              void* d_out, int out_size, void* d_ws, size_t ws_size,
                              hipStream_t stream) {
  (void)in_sizes; (void)n_in; (void)out_size;
  // setup_inputs order: protein(0), drug(1), mask_prot(2), mask_drug(3),
  //                     Wq_p(4), Wk_p(5), Wv_p(6), Wq_d(7), Wk_d(8), Wv_d(9)
  const float* prot = (const float*)d_in[0];
  const float* drug = (const float*)d_in[1];
  const float* wvp  = (const float*)d_in[6];
  const float* wvd  = (const float*)d_in[9];
  float* out = (float*)d_out;

  const size_t needX = (size_t)NROWS * KTOT * sizeof(u16);   // 128 MB
  const size_t needW = (size_t)HID * KTOT * sizeof(u16);     //  16 MB
  if (ws_size >= needX + needW) {
    u16* xb = (u16*)d_ws;
    u16* wt = xb + (size_t)NROWS * KTOT;
    convert_x<<<dim3((NROWS * (KTOT / 8)) / 256), 256, 0, stream>>>(prot, drug, xb);
    convert_w<<<dim3(HID / BN, KT / 2), 256, 0, stream>>>(wvp, wvd, wt);
    gemm_bf16<<<dim3(HID / BN, NROWS / BM), 256, 0, stream>>>(xb, wt, out);
  } else {
    can_fused_gemm<<<dim3(HID / BN, NROWS / BM), 256, 0, stream>>>(prot, drug, wvp, wvd, out);
  }
}
